// MultiHeadSelfAttention_13073880449469
// MI455X (gfx1250) — matrix operand order repaired and checked by reading, not compiled, not run on hardware
//
#include <hip/hip_runtime.h>
#include <hip/hip_bf16.h>

// ---------------------------------------------------------------------------
// MultiHeadSelfAttention for MI455X (gfx1250, wave32, WMMA bf16 + async-LDS)
//   B=4, C=256, H=W=48 -> N=2304, HEADS=8, DHEAD=32
//   kernel 1: qkv = Wqkv*x + b -> Q[n,d]*(scale*log2e) bf16, K[n,d], V^T[d,n]
//   kernel 2: flash attention; K/V chunks staged to LDS with
//             global_load_async_to_lds_b128 (ASYNCcnt), double buffered;
//             S^T = K Q^T, O^T = V^T P^T, fp32 accum, base-2 online softmax
//   kernel 3: out = W0*AO + b0 (fp32 out)
// ---------------------------------------------------------------------------

#define BATCH 4
#define CCH   256
#define NPIX  2304
#define HEADS 8
#define DHEAD 32
#define SEG   ((size_t)BATCH * HEADS * NPIX * DHEAD)   // elems per ws segment

typedef __attribute__((ext_vector_type(16))) __bf16         v16bf;
typedef __attribute__((ext_vector_type(2)))  __bf16         v2bf;
typedef __attribute__((ext_vector_type(8)))  float          v8f;
typedef __attribute__((ext_vector_type(8)))  unsigned int   v8u;
typedef __attribute__((ext_vector_type(16))) unsigned short v16us;
typedef __attribute__((ext_vector_type(8)))  unsigned short v8us;

static __device__ __forceinline__ unsigned short f2bf(float f) {
  unsigned int u = __builtin_bit_cast(unsigned int, f);
  u += 0x7FFFu + ((u >> 16) & 1u);   // round-to-nearest-even
  return (unsigned short)(u >> 16);
}
static __device__ __forceinline__ unsigned int pk2(float lo, float hi) {
#if __has_builtin(__builtin_amdgcn_cvt_pk_bf16_f32)
  v2bf t = __builtin_amdgcn_cvt_pk_bf16_f32(lo, hi);   // v_cvt_pk_bf16_f32
  return __builtin_bit_cast(unsigned int, t);
#else
  return (unsigned int)f2bf(lo) | ((unsigned int)f2bf(hi) << 16);
#endif
}
static __device__ __forceinline__ float exp2_fast(float x) {
  return __builtin_amdgcn_exp2f(x);                    // v_exp_f32
}

static __device__ __forceinline__ v8f wmma_bf16_u(v8u a, v8u b, v8f c) {
  return __builtin_amdgcn_wmma_f32_16x16x32_bf16(
      false, __builtin_bit_cast(v16bf, a),
      false, __builtin_bit_cast(v16bf, b),
      (short)0, c, false, false);
}
static __device__ __forceinline__ v8f wmma_bf16_s(v16us a, v16us b, v8f c) {
  return __builtin_amdgcn_wmma_f32_16x16x32_bf16(
      false, __builtin_bit_cast(v16bf, a),
      false, __builtin_bit_cast(v16bf, b),
      (short)0, c, false, false);
}

// --- CDNA5 async global->LDS copy (ASYNCcnt) -------------------------------
static __device__ __forceinline__ void async_copy_b128(unsigned lds_off,
                                                       const void* gptr) {
  asm volatile("global_load_async_to_lds_b128 %0, %1, off"
               :: "v"(lds_off),
                  "v"((unsigned long long)(size_t)gptr)
               : "memory");
}
static __device__ __forceinline__ void wait_async0() {
  asm volatile("s_wait_asynccnt 0" ::: "memory");
}

// ---------------------------------------------------------------------------
// Kernel 1: QKV projection. One 16x16 output tile per wave, K=256 in 8 steps.
// Output channel o in [0,768): o = d*24 + k*8 + m  (per reference rearrange).
// Branchless scatter: Q/K/V^T are consecutive segments of `ws`.
// ---------------------------------------------------------------------------
__global__ void __launch_bounds__(256)
qkv_kernel(const float* __restrict__ x, const float* __restrict__ Wqkv,
           const float* __restrict__ bqkv, unsigned short* __restrict__ ws) {
  const int lane = threadIdx.x & 31;
  const int wave = threadIdx.x >> 5;
  const int tile = blockIdx.x * 8 + wave;      // 27648 tiles total
  const int b    = tile / (48 * 144);
  const int rem  = tile % (48 * 144);
  const int ot   = rem / 144;
  const int pt   = rem % 144;
  const int col  = lane & 15;
  const int half = lane >> 4;
  const int p    = pt * 16 + col;
  const int arow = ot * 16 + col;
  const float* xb = x + (size_t)b * CCH * NPIX;

  v8f acc = {};
  for (int kc = 0; kc < 8; ++kc) {
    const float* wrow = Wqkv + (size_t)arow * CCH + kc * 32 + half * 16;
    v8u a;
#pragma unroll
    for (int w = 0; w < 4; ++w) {
      a[w]     = pk2(wrow[2 * w],      wrow[2 * w + 1]);
      a[4 + w] = pk2(wrow[8 + 2 * w], wrow[8 + 2 * w + 1]);
    }
    const float* xc = xb + (size_t)(kc * 32 + half * 16) * NPIX + p;
    v8u bfr;
#pragma unroll
    for (int w = 0; w < 8; ++w)
      bfr[w] = pk2(xc[(size_t)(2 * w) * NPIX], xc[(size_t)(2 * w + 1) * NPIX]);
    acc = wmma_bf16_u(a, bfr, acc);
  }

  // 32^-0.5 * log2(e): softmax runs in base 2 downstream
  const float qscale = 0.25503473662433574f;
#pragma unroll
  for (int r = 0; r < 8; ++r) {
    const int o = ot * 16 + (half ? 8 + r : r);
    float v = acc[r] + bqkv[o];
    const int d  = o / 24, rr = o - d * 24;
    const int kk = rr >> 3, mm = rr & 7;
    const int bm = b * 8 + mm;
    const size_t idx_qk = ((size_t)bm * NPIX + p) * DHEAD + d;  // Q/K [n,d]
    const size_t idx_v  = ((size_t)bm * DHEAD + d) * NPIX + p;  // V^T [d,n]
    v *= (kk == 0) ? qscale : 1.0f;
    ws[(size_t)kk * SEG + ((kk == 2) ? idx_v : idx_qk)] = f2bf(v);
  }
}

// ---------------------------------------------------------------------------
// Kernel 2: flash attention. Block = 4 waves (same b,m), wave = 16 queries.
// K/V 32-chunk staged in LDS via async copies, double buffered.
// LDS tile: 32 rows x 80B (padded; stride 20 banks -> conflict-free).
// ---------------------------------------------------------------------------
#define LROW 40                                // ushorts per LDS row (80 B)

__global__ void __launch_bounds__(128)
attn_kernel(const unsigned short* __restrict__ Qws,
            const unsigned short* __restrict__ Kws,
            const unsigned short* __restrict__ Vtws,
            unsigned short* __restrict__ AO) {
  const int lane = threadIdx.x & 31;
  const int wave = threadIdx.x >> 5;
  const int bm   = blockIdx.x;                 // 32
  const int q0   = (blockIdx.y * 4 + wave) * 16;
  const int col  = lane & 15;
  const int half = lane >> 4;

  __shared__ alignas(16) unsigned short sm[2][2][32 * LROW]; // [buf][K/V][..]

  const unsigned short* Qb = Qws  + (size_t)bm * NPIX * DHEAD;
  const char* Kg = (const char*)(Kws  + (size_t)bm * NPIX * DHEAD);
  const char* Vg = (const char*)(Vtws + (size_t)bm * DHEAD * NPIX);

  // Q^T B-fragment (persistent): lane -> query q0+col, d = half*8 + t (t<8), 16 + half*8 + (t-8) (t>=8)
  v16us qt;
  {
    const v8us* qv = (const v8us*)(Qb + (size_t)(q0 + col) * DHEAD + half * 8);
    const v8us lo = qv[0], hi = qv[2];
#pragma unroll
    for (int t = 0; t < 8; ++t) { qt[t] = lo[t]; qt[8 + t] = hi[t]; }
  }

  // Async stage of one 32-key chunk: wave0 -> K tile (rows = keys),
  // wave1 -> V^T tile (rows = d). 4 issues x 512B per wave.
  auto stage = [&](int buf, int k0) {
    if (wave < 2) {
      const char*  gbase = (wave == 0) ? (Kg + (size_t)k0 * 64)
                                       : (Vg + (size_t)k0 * 2);
      const size_t grs   = (wave == 0) ? (size_t)64 : (size_t)NPIX * 2;
      const unsigned lbase = (unsigned)(size_t)&sm[buf][wave][0];
      const int row0 = lane >> 2;              // 8 rows per issue
      const int coff = (lane & 3) * 16;        // 4 x 16B per row
#pragma unroll
      for (int i = 0; i < 4; ++i) {
        const int row = i * 8 + row0;
        async_copy_b128(lbase + (unsigned)(row * (LROW * 2) + coff),
                        gbase + (size_t)row * grs + coff);
      }
    }
  };

  v8f o0 = {}, o1 = {};
  float mrun = -3.0e38f, lrun = 0.0f;

  stage(0, 0);
  for (int k0 = 0, it = 0; k0 < NPIX; k0 += 32, ++it) {
    const int buf = it & 1;
    if (wave < 2) wait_async0();               // staging waves drain ASYNCcnt
    __syncthreads();                           // chunk visible to all waves
    if (k0 + 32 < NPIX) stage(buf ^ 1, k0 + 32);

    const unsigned short* Kl = &sm[buf][0][0];
    const unsigned short* Vl = &sm[buf][1][0];

    // --- K A-fragments from LDS: rows = keys, K-dim = d -------------------
    v16us kf0, kf1;
    {
      const v8us* a0 = (const v8us*)(Kl + col * LROW + half * 8);
      const v8us* a1 = (const v8us*)(Kl + (16 + col) * LROW + half * 8);
      const v8us x0 = a0[0], x1 = a0[2], y0 = a1[0], y1 = a1[2];
#pragma unroll
      for (int t = 0; t < 8; ++t) {
        kf0[t] = x0[t]; kf0[8 + t] = x1[t];
        kf1[t] = y0[t]; kf1[8 + t] = y1[t];
      }
    }
    const v8f z = {};
    v8f s0 = wmma_bf16_s(kf0, qt, z);          // S^T (log2-scaled) keys lo 16
    v8f s1 = wmma_bf16_s(kf1, qt, z);          // S^T keys hi 16

    // --- base-2 online softmax over key dim -------------------------------
    float mx = s0[0];
#pragma unroll
    for (int r = 0; r < 8; ++r) { mx = fmaxf(mx, s0[r]); mx = fmaxf(mx, s1[r]); }
    mx = fmaxf(mx, __shfl_xor(mx, 16, 32));
    const float mnew  = fmaxf(mrun, mx);
    const float alpha = exp2_fast(mrun - mnew);
    float ls = 0.0f;
#pragma unroll
    for (int r = 0; r < 8; ++r) {
      s0[r] = exp2_fast(s0[r] - mnew); ls += s0[r];
      s1[r] = exp2_fast(s1[r] - mnew); ls += s1[r];
    }
    ls   += __shfl_xor(ls, 16, 32);
    lrun  = lrun * alpha + ls;
    mrun  = mnew;
#pragma unroll
    for (int r = 0; r < 8; ++r) { o0[r] *= alpha; o1[r] *= alpha; }

    // --- rebuild P^T as B-fragment (C-layout -> B-layout via shfl_xor 16) -
    float e0[8], e1[8];
#pragma unroll
    for (int j = 0; j < 8; ++j) {
      const float oLo = __shfl_xor(s0[j], 16, 32);
      const float oHi = __shfl_xor(s1[j], 16, 32);
      e0[j] = (half == 0) ? s0[j] : oHi;       // elems 0..7  (keys j / 16+j)
      e1[j] = (half == 0) ? oLo   : s1[j];     // elems 8..15 (keys 8+j / 24+j)
    }
    v8u pf;
#pragma unroll
    for (int w = 0; w < 4; ++w) {
      pf[w]     = pk2(e0[2 * w], e0[2 * w + 1]);
      pf[4 + w] = pk2(e1[2 * w], e1[2 * w + 1]);
    }

    // --- V^T A-fragments from LDS: rows = d, K-dim = keys -----------------
    v16us vf0, vf1;
    {
      const v8us* a0 = (const v8us*)(Vl + col * LROW + half * 16);
      const v8us* a1 = (const v8us*)(Vl + (16 + col) * LROW + half * 16);
      const v8us x0 = a0[0], x1 = a0[1], y0 = a1[0], y1 = a1[1];
#pragma unroll
      for (int t = 0; t < 8; ++t) {
        vf0[t] = x0[t]; vf0[8 + t] = x1[t];
        vf1[t] = y0[t]; vf1[8 + t] = y1[t];
      }
    }
    o0 = wmma_bf16_s(vf0, __builtin_bit_cast(v16us, pf), o0); // O^T d 0..15
    o1 = wmma_bf16_s(vf1, __builtin_bit_cast(v16us, pf), o1); // O^T d 16..31
  }

  // --- normalize, store AO[b][m*32+d][n] as bf16 --------------------------
  const float rinv = 1.0f / lrun;
  const int b = bm >> 3, m = bm & 7;
  const size_t base = ((size_t)b * CCH + m * DHEAD) * NPIX + (q0 + col);
#pragma unroll
  for (int r = 0; r < 8; ++r) {
    const int d = half ? 8 + r : r;
    AO[base + (size_t)d * NPIX]        = f2bf(o0[r] * rinv);
    AO[base + (size_t)(16 + d) * NPIX] = f2bf(o1[r] * rinv);
  }
}

// ---------------------------------------------------------------------------
// Kernel 3: output projection, fp32 result to d_out ([b, C, H, W] = [b,o,p]).
// ---------------------------------------------------------------------------
__global__ void __launch_bounds__(256)
proj_kernel(const unsigned short* __restrict__ AO,
            const float* __restrict__ W0, const float* __restrict__ b0,
            float* __restrict__ out) {
  const int lane = threadIdx.x & 31;
  const int wave = threadIdx.x >> 5;
  const int tile = blockIdx.x * 8 + wave;      // 9216 tiles total
  const int b    = tile / (16 * 144);
  const int rem  = tile % (16 * 144);
  const int ot   = rem / 144;
  const int pt   = rem % 144;
  const int col  = lane & 15;
  const int half = lane >> 4;
  const int p    = pt * 16 + col;
  const int arow = ot * 16 + col;
  const unsigned short* aob = AO + (size_t)b * CCH * NPIX;

  v8f acc = {};
  for (int kc = 0; kc < 8; ++kc) {
    const float* wrow = W0 + (size_t)arow * CCH + kc * 32 + half * 16;
    v8u a;
#pragma unroll
    for (int w = 0; w < 4; ++w) {
      a[w]     = pk2(wrow[2 * w],      wrow[2 * w + 1]);
      a[4 + w] = pk2(wrow[8 + 2 * w], wrow[8 + 2 * w + 1]);
    }
    const unsigned short* xc = aob + (size_t)(kc * 32 + half * 16) * NPIX + p;
    v8u bfr;
#pragma unroll
    for (int w = 0; w < 8; ++w)
      bfr[w] = (unsigned int)xc[(size_t)(2 * w) * NPIX] |
               ((unsigned int)xc[(size_t)(2 * w + 1) * NPIX] << 16);
    acc = wmma_bf16_u(a, bfr, acc);
  }
#pragma unroll
  for (int r = 0; r < 8; ++r) {
    const int o = ot * 16 + (half ? 8 + r : r);
    out[((size_t)b * CCH + o) * NPIX + p] = acc[r] + b0[o];
  }
}

// ---------------------------------------------------------------------------
extern "C" void kernel_launch(void* const* d_in, const int* in_sizes, int n_in,
                              void* d_out, int out_size, void* d_ws, size_t ws_size,
                              hipStream_t stream) {
  const float* x    = (const float*)d_in[0];
  const float* Wqkv = (const float*)d_in[1];
  const float* bqkv = (const float*)d_in[2];
  const float* W0   = (const float*)d_in[3];
  const float* b0   = (const float*)d_in[4];
  float* out = (float*)d_out;

  // bf16 workspace: Q | K | V^T | AO — each SEG = 2,359,296 elems (4.5 MiB)
  unsigned short* ws  = (unsigned short*)d_ws;
  unsigned short* Qws = ws;
  unsigned short* Kws = ws + SEG;
  unsigned short* Vt  = ws + 2 * SEG;
  unsigned short* AO  = ws + 3 * SEG;

  qkv_kernel <<<3456, 256, 0, stream>>>(x, Wqkv, bqkv, ws);
  attn_kernel<<<dim3(32, 36), 128, 0, stream>>>(Qws, Kws, Vt, AO);
  proj_kernel<<<1152, 256, 0, stream>>>(AO, W0, b0, out);
}
